// MultiGCNLayer_36498632081513
// MI455X (gfx1250) — matrix operand order repaired and checked by reading, not compiled, not run on hardware
//
#include <hip/hip_runtime.h>
#include <hip/hip_bf16.h>

typedef __attribute__((ext_vector_type(16))) __bf16 v16bf;
typedef __attribute__((ext_vector_type(8)))  float  v8f;

#define HDIM   128
#define TILE_M 256
#define LDT    136   // padded bf16 row stride (272B = 68 dwords -> conflict-free)

#if defined(__has_builtin)
#if __has_builtin(__builtin_amdgcn_global_load_async_to_lds_b128) && \
    __has_builtin(__builtin_amdgcn_s_wait_asynccnt)
#define USE_ASYNC 1
#endif
#endif

// ---------------------------------------------------------------------------
// Async weight staging: global f32 -> LDS f32 via GLOBAL_LOAD_ASYNC_TO_LDS_B128
// (ASYNCcnt-tracked; overlapped with the previous GEMM stage).
// Builtin signature (from hipcc diagnostic): (int32x4 __device__*,
// int32x4 __shared__*, imm int offset, imm int cpol).
// ---------------------------------------------------------------------------
#if USE_ASYNC
typedef __attribute__((__vector_size__(16))) int i32x4;
typedef __attribute__((address_space(1))) i32x4 g_i32x4;
typedef __attribute__((address_space(3))) i32x4 l_i32x4;

__device__ __forceinline__ void async_copy_w(float* __restrict__ ldsDst,
                                             const float* __restrict__ gSrc,
                                             int tid) {
  const float* g = gSrc + tid * 4;   // 16B per lane per issue
  float* l = ldsDst + tid * 4;
#pragma unroll
  for (int i = 0; i < 16; ++i) {     // 16 sweeps * 256 thr * 4 f32 = 16384 f32
    __builtin_amdgcn_global_load_async_to_lds_b128(
        (g_i32x4*)(g + i * 1024), (l_i32x4*)(l + i * 1024), 0, 0);
  }
}
#define STAGE_ISSUE_W(gp)  async_copy_w(sWs, (gp), tid)
#define STAGE_WAIT_W()     __builtin_amdgcn_s_wait_asynccnt(0)
#define STAGE_W_SRC(gp)    ((const float*)sWs)
#else
#define STAGE_ISSUE_W(gp)  ((void)0)
#define STAGE_WAIT_W()     ((void)0)
#define STAGE_W_SRC(gp)    (gp)
#endif

// ---------------------------------------------------------------------------
// LDS loaders: f32 (global or LDS staging) -> bf16 LDS
// ---------------------------------------------------------------------------
__device__ __forceinline__ void load_tile_bf16(__bf16* __restrict__ dst,
                                               const float* __restrict__ src,
                                               int tid) {
  const float4* s4 = (const float4*)src;
  for (int idx = tid; idx < TILE_M * (HDIM / 4); idx += 256) {
    float4 v = s4[idx];
    int r = idx >> 5;
    int c = (idx & 31) << 2;
    __bf16* d = dst + r * LDT + c;
    d[0] = (__bf16)v.x; d[1] = (__bf16)v.y; d[2] = (__bf16)v.z; d[3] = (__bf16)v.w;
  }
}

// 128x128 W[k][n] (row-major f32) -> LDS transposed bf16: sWt[n*LDT + k]
__device__ __forceinline__ void load_weight_t(__bf16* __restrict__ dst,
                                              const float* __restrict__ src,
                                              int tid) {
  const float4* s4 = (const float4*)src;
  for (int idx = tid; idx < HDIM * (HDIM / 4); idx += 256) {
    float4 v = s4[idx];
    int k = idx >> 5;
    int n = (idx & 31) << 2;
    dst[(n + 0) * LDT + k] = (__bf16)v.x;
    dst[(n + 1) * LDT + k] = (__bf16)v.y;
    dst[(n + 2) * LDT + k] = (__bf16)v.z;
    dst[(n + 3) * LDT + k] = (__bf16)v.w;
  }
}

// ---------------------------------------------------------------------------
// Per-wave 32x128 stripe of (256x128)x(128x128) bf16 GEMM, f32 accumulate.
// B fragments are double-buffered to break the WAR hazard between in-flight
// WMMA reads of B and the next ds_load.
// ---------------------------------------------------------------------------
__device__ __forceinline__ void wave_gemm_256x128(const __bf16* __restrict__ sA,
                                                  const __bf16* __restrict__ sWt,
                                                  v8f acc[2][8],
                                                  int lane, int waveRow) {
  v8f zero;
#pragma unroll
  for (int i = 0; i < 8; ++i) zero[i] = 0.0f;
#pragma unroll
  for (int mt = 0; mt < 2; ++mt)
#pragma unroll
    for (int nt = 0; nt < 8; ++nt) acc[mt][nt] = zero;

  const int m  = lane & 15;
  const int kh = (lane >> 4) << 3;   // A: lanes 16..31 hold K+8 / K+24 halves
  const int n  = lane & 15;
  const int kb = (lane >> 4) << 3;   // B: lanes 16..31 hold K+8 / K+24 halves

  union BU { v16bf v; unsigned u[8]; };

  for (int ks = 0; ks < 4; ++ks) {
    const int k0 = ks * 32;
    union { v16bf v; unsigned u[8]; } af[2];
#pragma unroll
    for (int mt = 0; mt < 2; ++mt) {
      const __bf16* arow = sA + (waveRow + mt * 16 + m) * LDT;
      const unsigned* p0 = (const unsigned*)(arow + k0 + kh);
      const unsigned* p1 = (const unsigned*)(arow + k0 + 16 + kh);
#pragma unroll
      for (int v = 0; v < 4; ++v) { af[mt].u[v] = p0[v]; af[mt].u[v + 4] = p1[v]; }
    }

    auto loadB = [&](BU& d, int nt) {
      const unsigned* wp = (const unsigned*)(sWt + (nt * 16 + n) * LDT + k0 + kb);
#pragma unroll
      for (int v = 0; v < 8; ++v) d.u[v] = wp[v < 4 ? v : v + 4];
    };

    BU bcur, bnxt;
    loadB(bcur, 0);
#pragma unroll
    for (int nt = 0; nt < 8; ++nt) {
      if (nt < 7) loadB(bnxt, nt + 1);          // prefetch next B fragment
      acc[0][nt] = __builtin_amdgcn_wmma_f32_16x16x32_bf16(
          false, af[0].v, false, bcur.v, (short)0, acc[0][nt], false, false);
      acc[1][nt] = __builtin_amdgcn_wmma_f32_16x16x32_bf16(
          false, af[1].v, false, bcur.v, (short)0, acc[1][nt], false, false);
      if (nt < 7) bcur = bnxt;                  // renamed away by full unroll
    }
  }
}

// Accumulator element mapping (f32 16x16 C/D): row = rb + r, col = lane&15
#define ACC_FOREACH(body)                                         \
  {                                                               \
    const int rb = (lane >> 4) << 3;                              \
    const int cl = lane & 15;                                     \
    _Pragma("unroll")                                             \
    for (int mt = 0; mt < 2; ++mt)                                \
      _Pragma("unroll")                                           \
      for (int nt = 0; nt < 8; ++nt) {                            \
        const int col = nt * 16 + cl;                             \
        _Pragma("unroll")                                         \
        for (int r = 0; r < 8; ++r) {                             \
          const int row = waveRow + mt * 16 + rb + r;             \
          body                                                    \
        }                                                         \
      }                                                           \
  }

// LDS carve shared by the GEMM kernels (sWs present even in fallback builds
// so host/device shared-mem sizes agree).
#define CARVE_SMEM()                                              \
  extern __shared__ char smem[];                                  \
  __bf16* sA  = (__bf16*)smem;                                    \
  __bf16* sH  = sA + TILE_M * LDT;                                \
  __bf16* sWt = sH + TILE_M * LDT;                                \
  float*  sWs = (float*)(sWt + HDIM * LDT);   /* f32 staging  */  \
  float* sBias = sWs + HDIM * HDIM;                               \
  float* sAcc  = sBias + HDIM;                                    \
  float* sAw   = sAcc + HDIM;                                     \
  (void)sWs; (void)sAcc; (void)sAw;

// ---------------------------------------------------------------------------
// Kernel 1: node_new = node_embed + relu(node_embed@W1+b1)@W2 + b2
// ---------------------------------------------------------------------------
__global__ void __launch_bounds__(256)
node_mlp_kernel(const float* __restrict__ x,
                const float* __restrict__ W1, const float* __restrict__ b1,
                const float* __restrict__ W2, const float* __restrict__ b2,
                float* __restrict__ out) {
  CARVE_SMEM();
  const int tid = threadIdx.x, lane = tid & 31, waveRow = (tid >> 5) * 32;
  const float* X = x + (size_t)blockIdx.x * TILE_M * HDIM;
  float* O = out + (size_t)blockIdx.x * TILE_M * HDIM;

  STAGE_ISSUE_W(W1);
  load_tile_bf16(sA, X, tid);
  STAGE_WAIT_W();
  __syncthreads();
  load_weight_t(sWt, STAGE_W_SRC(W1), tid);
  if (tid < HDIM) sBias[tid] = b1[tid];
  __syncthreads();

  STAGE_ISSUE_W(W2);
  v8f acc[2][8];
  wave_gemm_256x128(sA, sWt, acc, lane, waveRow);
  ACC_FOREACH({
    float v = acc[mt][nt][r] + sBias[col];
    sH[row * LDT + col] = (__bf16)(v > 0.f ? v : 0.f);
  })
  STAGE_WAIT_W();
  __syncthreads();
  load_weight_t(sWt, STAGE_W_SRC(W2), tid);
  if (tid < HDIM) sBias[tid] = b2[tid];
  __syncthreads();

  wave_gemm_256x128(sH, sWt, acc, lane, waveRow);
  ACC_FOREACH({
    O[row * HDIM + col] = X[row * HDIM + col] + acc[mt][nt][r] + sBias[col];
  })
}

// ---------------------------------------------------------------------------
// Kernel 2: fused edge pipeline, one workgroup per (b,i) 256x128 tile.
// edge MLP -> *node_new -> dist MLP -> A_dist-weighted j-reduce -> rel MLP.
// Attention softmax is over a singleton axis (== 1), so Aw == A_dist.
// Next-stage weight panels stream in via async-to-LDS under the current GEMM.
// ---------------------------------------------------------------------------
__global__ void __launch_bounds__(256)
gcn_edge_fused_kernel(const float* __restrict__ edge_fea,
                      const float* __restrict__ A_dist,
                      const float* __restrict__ eW1, const float* __restrict__ eb1,
                      const float* __restrict__ eW2, const float* __restrict__ eb2,
                      const float* __restrict__ dW1, const float* __restrict__ db1,
                      const float* __restrict__ dW2, const float* __restrict__ db2,
                      const float* __restrict__ rW1, const float* __restrict__ rb1,
                      const float* __restrict__ rW2, const float* __restrict__ rb2,
                      const float* __restrict__ node_new,
                      float* __restrict__ node_pre,
                      float* __restrict__ out_edge) {
  CARVE_SMEM();
  const int tid = threadIdx.x, lane = tid & 31, waveRow = (tid >> 5) * 32;
  const int bi = blockIdx.x;                   // b*N + i
  const int b  = bi >> 8;                      // N = 256

  const float* Erow  = edge_fea + (size_t)bi * 256 * HDIM;
  const float* nodeB = node_new + (size_t)b * 256 * HDIM;
  float* outRow = out_edge + (size_t)bi * 256 * HDIM;

  if (tid < HDIM) sAcc[tid] = 0.f;
  sAw[tid] = A_dist[(size_t)bi * 256 + tid];

  // ---- stage 1: hidden = relu(E @ eW1 + eb1)
  STAGE_ISSUE_W(eW1);
  load_tile_bf16(sA, Erow, tid);
  STAGE_WAIT_W();
  __syncthreads();
  load_weight_t(sWt, STAGE_W_SRC(eW1), tid);
  if (tid < HDIM) sBias[tid] = eb1[tid];
  __syncthreads();

  STAGE_ISSUE_W(eW2);
  v8f acc[2][8];
  wave_gemm_256x128(sA, sWt, acc, lane, waveRow);
  ACC_FOREACH({
    float v = acc[mt][nt][r] + sBias[col];
    sH[row * LDT + col] = (__bf16)(v > 0.f ? v : 0.f);
  })
  STAGE_WAIT_W();
  __syncthreads();

  // ---- stage 2: ean = (E + hidden @ eW2 + eb2) * node_new[b,j,:] -> sA
  load_weight_t(sWt, STAGE_W_SRC(eW2), tid);
  if (tid < HDIM) sBias[tid] = eb2[tid];
  __syncthreads();

  STAGE_ISSUE_W(dW1);
  wave_gemm_256x128(sH, sWt, acc, lane, waveRow);
  ACC_FOREACH({
    float e = Erow[row * HDIM + col];
    float v = (e + acc[mt][nt][r] + sBias[col]) * nodeB[row * HDIM + col];
    sA[row * LDT + col] = (__bf16)v;
  })
  STAGE_WAIT_W();
  __syncthreads();

  // ---- stage 3: h2 = relu(ean @ dW1 + db1)
  load_weight_t(sWt, STAGE_W_SRC(dW1), tid);
  if (tid < HDIM) sBias[tid] = db1[tid];
  __syncthreads();

  STAGE_ISSUE_W(dW2);
  wave_gemm_256x128(sA, sWt, acc, lane, waveRow);
  ACC_FOREACH({
    float v = acc[mt][nt][r] + sBias[col];
    sH[row * LDT + col] = (__bf16)(v > 0.f ? v : 0.f);
  })
  STAGE_WAIT_W();
  __syncthreads();

  // ---- stage 4: msg = ean + h2 @ dW2 + db2 ; node += A_dist[j]*msg
  load_weight_t(sWt, STAGE_W_SRC(dW2), tid);
  if (tid < HDIM) sBias[tid] = db2[tid];
  __syncthreads();

  STAGE_ISSUE_W(rW1);
  wave_gemm_256x128(sH, sWt, acc, lane, waveRow);
  {
    const int rb = (lane >> 4) << 3;
    const int cl = lane & 15;
#pragma unroll
    for (int mt = 0; mt < 2; ++mt)
#pragma unroll
      for (int nt = 0; nt < 8; ++nt) {
        const int col = nt * 16 + cl;
        float part = 0.f;
#pragma unroll
        for (int r = 0; r < 8; ++r) {
          const int row = waveRow + mt * 16 + rb + r;
          float msg = (float)sA[row * LDT + col] + acc[mt][nt][r] + sBias[col];
          part += sAw[row] * msg;
        }
        atomicAdd(&sAcc[col], part);   // ds_add_f32
      }
  }
  STAGE_WAIT_W();
  __syncthreads();
  if (tid < HDIM) node_pre[(size_t)bi * HDIM + tid] = sAcc[tid];

  // ---- stage 5: h3 = relu(E @ rW1 + rb1)   (tile reload hits L2)
  load_tile_bf16(sA, Erow, tid);
  load_weight_t(sWt, STAGE_W_SRC(rW1), tid);
  if (tid < HDIM) sBias[tid] = rb1[tid];
  __syncthreads();

  STAGE_ISSUE_W(rW2);
  wave_gemm_256x128(sA, sWt, acc, lane, waveRow);
  ACC_FOREACH({
    float v = acc[mt][nt][r] + sBias[col];
    sH[row * LDT + col] = (__bf16)(v > 0.f ? v : 0.f);
  })
  STAGE_WAIT_W();
  __syncthreads();

  // ---- stage 6: out_edge = E + h3 @ rW2 + rb2
  load_weight_t(sWt, STAGE_W_SRC(rW2), tid);
  if (tid < HDIM) sBias[tid] = rb2[tid];
  __syncthreads();

  wave_gemm_256x128(sH, sWt, acc, lane, waveRow);
  ACC_FOREACH({
    outRow[row * HDIM + col] = Erow[row * HDIM + col] + acc[mt][nt][r] + sBias[col];
  })
}

// ---------------------------------------------------------------------------
// Kernel 3: BatchNorm over (B*N) rows per channel, then ReLU -> out_node
// ---------------------------------------------------------------------------
__global__ void __launch_bounds__(128)
bn_relu_kernel(const float* __restrict__ node_pre,
               const float* __restrict__ gamma, const float* __restrict__ beta,
               float* __restrict__ out) {
  const int h = threadIdx.x;
  float s = 0.f, s2 = 0.f;
  for (int r = 0; r < 1024; ++r) {
    float v = node_pre[r * HDIM + h];
    s += v; s2 += v * v;
  }
  const float mean = s * (1.f / 1024.f);
  const float var  = s2 * (1.f / 1024.f) - mean * mean;
  const float sc = rsqrtf(var + 1e-5f) * gamma[h];
  const float sh = beta[h] - mean * sc;
  for (int r = 0; r < 1024; ++r) {
    float v = node_pre[r * HDIM + h] * sc + sh;
    out[r * HDIM + h] = v > 0.f ? v : 0.f;
  }
}

// ---------------------------------------------------------------------------
extern "C" void kernel_launch(void* const* d_in, const int* in_sizes, int n_in,
                              void* d_out, int out_size, void* d_ws, size_t ws_size,
                              hipStream_t stream) {
  const float* node_embed = (const float*)d_in[0];
  const float* edge_fea   = (const float*)d_in[1];
  const float* A_dist     = (const float*)d_in[2];
  const float* nW1 = (const float*)d_in[3],  *nb1 = (const float*)d_in[4];
  const float* nW2 = (const float*)d_in[5],  *nb2 = (const float*)d_in[6];
  const float* eW1 = (const float*)d_in[7],  *eb1 = (const float*)d_in[8];
  const float* eW2 = (const float*)d_in[9],  *eb2 = (const float*)d_in[10];
  const float* dW1 = (const float*)d_in[11], *db1 = (const float*)d_in[12];
  const float* dW2 = (const float*)d_in[13], *db2 = (const float*)d_in[14];
  const float* rW1 = (const float*)d_in[15], *rb1 = (const float*)d_in[16];
  const float* rW2 = (const float*)d_in[17], *rb2 = (const float*)d_in[18];
  // d_in[19] (attention) is mathematically dead: softmax over a size-1 axis == 1
  const float* gamma = (const float*)d_in[20];
  const float* beta  = (const float*)d_in[21];

  float* out_node = (float*)d_out;                 // (B,N,H) f32
  float* out_edge = out_node + 4 * 256 * HDIM;     // (B,N,N,H) f32

  float* node_new = (float*)d_ws;                  // (B,N,H) f32
  float* node_pre = node_new + 4 * 256 * HDIM;     // (B,N,H) f32

  // bf16 tiles + bf16 weight panel + f32 async staging + f32 vectors
  const size_t smem = (size_t)(2 * TILE_M * LDT + HDIM * LDT) * 2
                    + (size_t)(HDIM * HDIM) * 4
                    + (size_t)(HDIM + HDIM + 256) * 4;

  node_mlp_kernel<<<4, 256, smem, stream>>>(node_embed, nW1, nb1, nW2, nb2, node_new);
  gcn_edge_fused_kernel<<<1024, 256, smem, stream>>>(
      edge_fea, A_dist, eW1, eb1, eW2, eb2, dW1, db1, dW2, db2,
      rW1, rb1, rW2, rb2, node_new, node_pre, out_edge);
  bn_relu_kernel<<<1, 128, 0, stream>>>(node_pre, gamma, beta, out_node);
}
